// BaseVectorQuantizer_38740605010386
// MI455X (gfx1250) — matrix operand order repaired and checked by reading, not compiled, not run on hardware
//
#include <hip/hip_runtime.h>
#include <hip/hip_bf16.h>
#include <stdint.h>

typedef __attribute__((ext_vector_type(16))) __bf16 v16bf;
typedef __attribute__((ext_vector_type(8)))  __bf16 v8bf;
typedef __attribute__((ext_vector_type(8)))  float  v8f;

#define LN_EPS 1e-5f
constexpr int NEMB  = 1024;   // codebook size
constexpr int DIM   = 256;    // embedding dim
constexpr int NHID  = 512;    // hidden dim
constexpr int NROWS = 32768;  // B*T = 32*1024

// ---------------- low-level helpers ----------------

__device__ __forceinline__ v8f vzero8() {
  v8f z;
#pragma unroll
  for (int i = 0; i < 8; ++i) z[i] = 0.0f;
  return z;
}

// 16B global -> LDS async copy (per-lane), tracked by ASYNCcnt.
__device__ __forceinline__ void async_ld16(uint32_t ldsoff, const void* g) {
  uint64_t ga = (uint64_t)g;
  asm volatile("global_load_async_to_lds_b128 %0, %1, off"
               :: "v"(ldsoff), "v"(ga) : "memory");
}
__device__ __forceinline__ void wait_async8() {  // <=8 outstanding (prev chunk done)
  asm volatile("s_wait_asynccnt 0x8" ::: "memory");
}
__device__ __forceinline__ void wait_async0() {  // all done
  asm volatile("s_wait_asynccnt 0x0" ::: "memory");
}
__device__ __forceinline__ uint32_t lds_off(const void* p) {
  return (uint32_t)(uintptr_t)p;   // LDS aperture: addr[31:0] == LDS byte offset
}

// A fragment (16x32 bf16): lane m = lane&15, half = lane>>4.
// VGPR0-3: K = kbase + 8*half + 0..7 ; VGPR4-7: K = kbase + 16 + 8*half + 0..7
__device__ __forceinline__ v16bf a_frag_from_f32(const float* __restrict__ rowp,
                                                 int kbase, int half) {
  const float* p0 = rowp + kbase + half * 8;
  const float* p1 = p0 + 16;
  float4 x0 = ((const float4*)p0)[0];
  float4 x1 = ((const float4*)p0)[1];
  float4 y0 = ((const float4*)p1)[0];
  float4 y1 = ((const float4*)p1)[1];
  v16bf r;
  r[0]  = (__bf16)x0.x; r[1]  = (__bf16)x0.y; r[2]  = (__bf16)x0.z; r[3]  = (__bf16)x0.w;
  r[4]  = (__bf16)x1.x; r[5]  = (__bf16)x1.y; r[6]  = (__bf16)x1.z; r[7]  = (__bf16)x1.w;
  r[8]  = (__bf16)y0.x; r[9]  = (__bf16)y0.y; r[10] = (__bf16)y0.z; r[11] = (__bf16)y0.w;
  r[12] = (__bf16)y1.x; r[13] = (__bf16)y1.y; r[14] = (__bf16)y1.z; r[15] = (__bf16)y1.w;
  return r;
}

__device__ __forceinline__ v16bf a_frag_bf16(const __bf16* __restrict__ rowp,
                                             int kbase, int half) {
  const v8bf* p0 = (const v8bf*)(rowp + kbase + half * 8);
  const v8bf* p1 = (const v8bf*)(rowp + kbase + half * 8 + 16);
  v8bf lo = *p0, hi = *p1;
  return __builtin_shufflevector(lo, hi, 0,1,2,3,4,5,6,7,8,9,10,11,12,13,14,15);
}

// B fragment (32x16 KxN): lane n = lane&15 holds column n; 32 contiguous K per
// column, elems 0-7: K=kbase+8*half+0..7, elems 8-15: K=kbase+16+8*half+0..7.
// colp points at a column's contiguous 32-value K-run (works for LDS or global).
__device__ __forceinline__ v16bf b_frag(const __bf16* colp, int kbase, int half) {
  const v8bf* p = (const v8bf*)(colp + kbase + half * 8);
  v8bf lo = p[0], hi = p[2];
  return __builtin_shufflevector(lo, hi, 0,1,2,3,4,5,6,7,8,9,10,11,12,13,14,15);
}

__device__ __forceinline__ v8f wmma_bf16(v16bf a, v16bf b, v8f c) {
  return __builtin_amdgcn_wmma_f32_16x16x32_bf16(false, a, false, b,
                                                 (short)0, c, false, false);
}

// ---------------- prep kernels ----------------

__global__ void k_cvt_w(const float* __restrict__ w, __bf16* __restrict__ wbf, int n) {
  int i = blockIdx.x * blockDim.x + threadIdx.x;
  if (i < n) wbf[i] = (__bf16)w[i];
}

__global__ __launch_bounds__(256) void k_prep_emb(const float* __restrict__ emb,
                                                  __bf16* __restrict__ embbf,
                                                  float* __restrict__ embn) {
  __shared__ float sm[256];
  int e = blockIdx.x, d = threadIdx.x;
  float v = emb[(size_t)e * DIM + d];
  embbf[(size_t)e * DIM + d] = (__bf16)v;
  sm[d] = v * v;
  __syncthreads();
  for (int s = 128; s > 0; s >>= 1) {
    if (d < s) sm[d] += sm[d + s];
    __syncthreads();
  }
  if (d == 0) embn[e] = sm[0];
}

// ---------------- kernel 1: proj GEMM + bias + LayerNorm -> flat (bf16) --------
// 4 waves/block, each wave computes 16 rows x 256 cols over K=512.
// W is staged through LDS: chunk = one K-slice (256 cols x 32 K = 16 KB),
// layout [n][32] so each column's K-run is contiguous. Double buffered + async.

__global__ __launch_bounds__(128) void k_proj_ln(const float* __restrict__ feat,
                                                 const __bf16* __restrict__ wbf,
                                                 const float* __restrict__ bproj,
                                                 const float* __restrict__ gamma,
                                                 const float* __restrict__ beta,
                                                 __bf16* __restrict__ flat) {
  __shared__ __bf16 wsh[2][256 * 32];   // 2 x 16 KB

  const int tid  = threadIdx.x;
  const int lane = tid & 31;
  const int wave = tid >> 5;
  const int l15  = lane & 15;
  const int half = lane >> 4;
  const int row0 = (blockIdx.x * 4 + wave) * 16;

  v8f acc[16];
#pragma unroll
  for (int nt = 0; nt < 16; ++nt) acc[nt] = vzero8();

  const float* rowp = feat + (size_t)(row0 + l15) * NHID;

  // stage(kt): copy W[:, kt*32 .. +31] -> buf, 1024 x 16B segments, 8/thread
  auto stage = [&](int kt, int b) {
#pragma unroll
    for (int i = 0; i < 8; ++i) {
      int idx = tid + i * 128;          // 0..1023
      int n = idx >> 2;                 // 0..255
      int s = idx & 3;                  // 0..3 (16B segs of 64B row)
      async_ld16(lds_off(&wsh[b][n * 32 + s * 8]),
                 wbf + (size_t)n * NHID + kt * 32 + s * 8);
    }
  };
  auto compute = [&](int kt, int b) {
    v16bf a = a_frag_from_f32(rowp, kt * 32, half);
#pragma unroll
    for (int nt = 0; nt < 16; ++nt) {
      v16bf bb = b_frag(&wsh[b][(nt * 16 + l15) * 32], 0, half);
      acc[nt] = wmma_bf16(a, bb, acc[nt]);
    }
  };

  stage(0, 0);
  for (int kt = 0; kt < NHID / 32 - 1; ++kt) {
    stage(kt + 1, (kt + 1) & 1);
    wait_async8();
    __syncthreads();
    compute(kt, kt & 1);
    __syncthreads();
  }
  wait_async0();
  __syncthreads();
  compute(NHID / 32 - 1, (NHID / 32 - 1) & 1);

  // bias add
#pragma unroll
  for (int nt = 0; nt < 16; ++nt) {
    float bi = bproj[nt * 16 + l15];
#pragma unroll
    for (int r = 0; r < 8; ++r) acc[nt][r] += bi;
  }

  // LayerNorm per row (row M = r + 8*half lives across one 16-lane group)
#pragma unroll
  for (int r = 0; r < 8; ++r) {
    float s = 0.f, s2 = 0.f;
#pragma unroll
    for (int nt = 0; nt < 16; ++nt) {
      float v = acc[nt][r];
      s += v; s2 += v * v;
    }
#pragma unroll
    for (int m = 1; m < 16; m <<= 1) {
      s  += __shfl_xor(s,  m, 32);
      s2 += __shfl_xor(s2, m, 32);
    }
    float mean = s * (1.0f / 256.0f);
    float var  = s2 * (1.0f / 256.0f) - mean * mean;
    float rstd = rsqrtf(var + LN_EPS);

    __bf16* dst = flat + (size_t)(row0 + r + half * 8) * DIM;
#pragma unroll
    for (int nt = 0; nt < 16; ++nt) {
      int n = nt * 16 + l15;
      float v = (acc[nt][r] - mean) * rstd * gamma[n] + beta[n];
      dst[n] = (__bf16)v;
    }
  }
}

// ---------------- kernel 2: distance GEMM + argmin + outputs ----------------
// dist(m,e) = ||emb_e||^2 - 2 * flat_m . emb_e  (row-constant ||flat||^2 dropped)
// Codebook staged through LDS: chunk = 32 codes x 256 dims = 16 KB, dbl buffered.

__global__ __launch_bounds__(128) void k_vq(const __bf16* __restrict__ flat,
                                            const __bf16* __restrict__ embbf,
                                            const float* __restrict__ embn,
                                            const float* __restrict__ embedding,
                                            float* __restrict__ outq,
                                            float* __restrict__ outi,
                                            float* __restrict__ oute) {
  __shared__ __bf16 esh[2][32 * 256];   // 2 x 16 KB
  __shared__ int sidx[4][16];

  const int tid  = threadIdx.x;
  const int lane = tid & 31;
  const int wave = tid >> 5;
  const int l15  = lane & 15;
  const int half = lane >> 4;
  const int row0 = (blockIdx.x * 4 + wave) * 16;

  // preload all A fragments for this 16-row tile (K=256 -> 8 fragments, 64 VGPRs)
  const __bf16* arow = flat + (size_t)(row0 + l15) * DIM;
  v16bf af[8];
#pragma unroll
  for (int kt = 0; kt < 8; ++kt) af[kt] = a_frag_bf16(arow, kt * 32, half);

  float minv[8];
  int   mini[8];
#pragma unroll
  for (int r = 0; r < 8; ++r) { minv[r] = 3.4e38f; mini[r] = 0; }

  // stage(c): copy codes [c*32 .. c*32+31] -> buf (1024 x 16B segs, 8/thread)
  auto stage = [&](int c, int b) {
#pragma unroll
    for (int i = 0; i < 8; ++i) {
      int idx = tid + i * 128;          // 0..1023
      int r = idx >> 5;                 // 0..31 (code row in chunk)
      int s = idx & 31;                 // 0..31 (16B segs of 512B row)
      async_ld16(lds_off(&esh[b][r * 256 + s * 8]),
                 embbf + (size_t)(c * 32 + r) * DIM + s * 8);
    }
  };
  auto compute = [&](int c, int b) {
#pragma unroll
    for (int t = 0; t < 2; ++t) {       // two 16-code tiles per chunk
      v8f acc = vzero8();
      const __bf16* colp = &esh[b][(t * 16 + l15) * 256];
#pragma unroll
      for (int kt = 0; kt < 8; ++kt) {
        v16bf bb = b_frag(colp, kt * 32, half);
        acc = wmma_bf16(af[kt], bb, acc);
      }
      int   eidx = c * 32 + t * 16 + l15;
      float en   = embn[eidx];
#pragma unroll
      for (int r = 0; r < 8; ++r) {
        float d = en - 2.0f * acc[r];
        if (d < minv[r]) { minv[r] = d; mini[r] = eidx; }
      }
    }
  };

  stage(0, 0);
  for (int c = 0; c < NEMB / 32 - 1; ++c) {
    stage(c + 1, (c + 1) & 1);
    wait_async8();
    __syncthreads();
    compute(c, c & 1);
    __syncthreads();
  }
  wait_async0();
  __syncthreads();
  compute(NEMB / 32 - 1, (NEMB / 32 - 1) & 1);

  // butterfly min-reduce across each 16-lane group (lowest index wins ties)
#pragma unroll
  for (int r = 0; r < 8; ++r) {
#pragma unroll
    for (int m = 1; m < 16; m <<= 1) {
      float ov = __shfl_xor(minv[r], m, 32);
      int   oi = __shfl_xor(mini[r], m, 32);
      if (ov < minv[r] || (ov == minv[r] && oi < mini[r])) { minv[r] = ov; mini[r] = oi; }
    }
  }

  if (l15 == 0) {
#pragma unroll
    for (int r = 0; r < 8; ++r) {
      int j = r + half * 8;
      sidx[wave][j] = mini[r];
      outi[row0 + j] = (float)mini[r];   // indices output (as float)
    }
  }
  __syncthreads();

  // quantized = embedding[argmin] (fp32 gather, wave-cooperative float4)
  for (int j = 0; j < 16; ++j) {
    int e = sidx[wave][j];
    const float4* src = (const float4*)(embedding + (size_t)e * DIM);
    float4* dst = (float4*)(outq + (size_t)(row0 + j) * DIM);
    for (int i = lane; i < DIM / 4; i += 32) dst[i] = src[i];
  }

  // one-hot encodings rows: zero-fill with the 1.0 fused into the same store
  for (int j = 0; j < 16; ++j) {
    int e = sidx[wave][j];
    int slot = e >> 2, pos = e & 3;
    float4* dst = (float4*)(oute + (size_t)(row0 + j) * NEMB);
    for (int i = lane; i < NEMB / 4; i += 32) {
      float4 v = make_float4(0.f, 0.f, 0.f, 0.f);
      if (i == slot) (&v.x)[pos] = 1.0f;
      dst[i] = v;
    }
  }
}

// ---------------- launch ----------------

extern "C" void kernel_launch(void* const* d_in, const int* in_sizes, int n_in,
                              void* d_out, int out_size, void* d_ws, size_t ws_size,
                              hipStream_t stream) {
  const float* feat  = (const float*)d_in[0];  // [32,1024,512]
  const float* wproj = (const float*)d_in[1];  // [256,512]
  const float* bproj = (const float*)d_in[2];  // [256]
  const float* gamma = (const float*)d_in[3];  // [256]
  const float* beta  = (const float*)d_in[4];  // [256]
  const float* emb   = (const float*)d_in[5];  // [1024,256]

  char* ws = (char*)d_ws;
  __bf16* wbf   = (__bf16*)(ws);                              // 256 KB
  __bf16* embbf = (__bf16*)(ws + 262144);                     // 512 KB
  float*  embn  = (float*) (ws + 262144 + 524288);            // 4 KB
  __bf16* flat  = (__bf16*)(ws + 262144 + 524288 + 4096);     // 16 MB

  float* outq = (float*)d_out;                    // [32768,256]
  float* outi = outq + (size_t)NROWS * DIM;       // [32768] (as float)
  float* oute = outi + NROWS;                     // [32768,1024]

  k_cvt_w   <<<(NHID * DIM + 255) / 256, 256, 0, stream>>>(wproj, wbf, NHID * DIM);
  k_prep_emb<<<NEMB, 256, 0, stream>>>(emb, embbf, embn);
  k_proj_ln <<<NROWS / 64, 128, 0, stream>>>(feat, wbf, bproj, gamma, beta, flat);
  k_vq      <<<NROWS / 64, 128, 0, stream>>>(flat, embbf, embn, emb, outq, outi, oute);
}
